// VisitEmbedding_24086176596074
// MI455X (gfx1250) — matrix operand order repaired and checked by reading, not compiled, not run on hardware
//
#include <hip/hip_runtime.h>

typedef unsigned short u16;
typedef __attribute__((ext_vector_type(16))) __bf16 v16bf;
typedef __attribute__((ext_vector_type(8)))  float  v8f;

#define Bn 32
#define Lq 512
#define Cd 1024
#define Md (Bn * Lq)

__device__ __forceinline__ u16 f32_to_bf16(float f) {
    unsigned int u = __float_as_uint(f);
    u += 0x7fffu + ((u >> 16) & 1u);          // round-to-nearest-even
    return (u16)(u >> 16);
}
__device__ __forceinline__ float bf16_to_f32(u16 h) {
    return __uint_as_float(((unsigned int)h) << 16);
}
// 2-bit reverse: A-fragment k-chunk order 0,2,1,3 -> each lane's 16 fragment
// elements are one contiguous 32B run at offset half*16.
__device__ __forceinline__ int chunk_perm(int g) {
    return ((g & 1) << 1) | ((g >> 1) & 1);
}
// Low 32 bits of a generic LDS pointer == LDS byte offset (ISA 10.2 aperture).
__device__ __forceinline__ unsigned lds_off(const void* p) {
    return (unsigned)(size_t)p;
}
// CDNA5 async DMA: LDS[vdst] = MEM[saddr + voffset], tracked by ASYNCcnt.
__device__ __forceinline__ void async_b128(unsigned lds, unsigned voff,
                                           const void* sbase) {
    asm volatile("global_load_async_to_lds_b128 %0, %1, %2"
                 :: "v"(lds), "v"(voff), "s"(sbase) : "memory");
}
__device__ __forceinline__ void wait_async0() {
    asm volatile("s_wait_asynccnt 0x0" ::: "memory");
}

__global__ void convert_f32_bf16(const float* __restrict__ in,
                                 u16* __restrict__ out, int n4) {
    int i = (blockIdx.x * blockDim.x + threadIdx.x) << 2;
    if (i < n4) {
        float4 f = *(const float4*)&in[i];
        u16 o[4] = {f32_to_bf16(f.x), f32_to_bf16(f.y),
                    f32_to_bf16(f.z), f32_to_bf16(f.w)};
        *(uint2*)&out[i] = *(const uint2*)o;
    }
}

// ---------------------------------------------------------------------------
// bf16 GEMM (M=16384, N=K=1024): 128 threads / 4 waves, tile 128x64, K slab
// 32, double-buffered LDS. DMA/loads for slab n+1 issue under slab n's WMMAs;
// one barrier per slab. VTRANS: store V pre-transposed [batch][col][kpos].
// ---------------------------------------------------------------------------
template <int VTRANS>
__global__ __launch_bounds__(128)
void gemm_bf16(const u16* __restrict__ A, const u16* __restrict__ Bm,
               u16* __restrict__ D) {
    __shared__ __attribute__((aligned(32))) u16 lsA[2][128 * 32];  // 16 KB
    __shared__ __attribute__((aligned(32))) u16 lsBt[2][64 * 32];  //  8 KB

    const int tid  = threadIdx.x;
    const int wave = tid >> 5, lane = tid & 31;
    const int half = lane >> 4, l16 = lane & 15;
    const int m0 = blockIdx.y * 128, n0 = blockIdx.x * 64;

    v8f acc[2][4];
    for (int r = 0; r < 2; ++r)
        for (int t = 0; t < 4; ++t)
            for (int e = 0; e < 8; ++e) acc[r][t][e] = 0.f;

    const int ar0 = tid >> 2, ag = tid & 3;
    const unsigned aLds[2] = {
        lds_off(&lsA[0][ar0 * 32 + (ag << 3)]),
        lds_off(&lsA[1][ar0 * 32 + (ag << 3)])};
    const unsigned aOff0 = (unsigned)((m0 + ar0) * (Cd * 2) + (ag << 4));
    const int bk0 = tid >> 3, bc = (tid & 7) << 3;

    auto fillA = [&](int kb, int buf) {
        const u16* Aslab = A + (kb << 5);
        #pragma unroll
        for (int it = 0; it < 4; ++it)
            async_b128(aLds[buf] + it * 2048, aOff0 + it * (32 * Cd * 2), Aslab);
    };
    auto loadB = [&](int kb, uint4 seg[2]) {
        const u16* Bslab = Bm + (size_t)(kb << 5) * Cd + n0;
        #pragma unroll
        for (int it = 0; it < 2; ++it)
            seg[it] = *(const uint4*)&Bslab[(size_t)(bk0 + it * 16) * Cd + bc];
    };
    auto storeB = [&](const uint4 seg[2], int buf) {
        #pragma unroll
        for (int it = 0; it < 2; ++it) {
            u16 tmp[8];
            *(uint4*)tmp = seg[it];
            #pragma unroll
            for (int e = 0; e < 8; ++e)
                lsBt[buf][(bc + e) * 32 + bk0 + it * 16] = tmp[e];
        }
    };

    uint4 segN[2];
    fillA(0, 0);
    loadB(0, segN);
    storeB(segN, 0);

    for (int kb = 0; kb < Cd / 32; ++kb) {
        const int cur = kb & 1;
        wait_async0();          // own DMA for buf[cur] done
        __syncthreads();        // everyone done reading buf[cur^1], buf[cur] ready
        if (kb + 1 < Cd / 32) {
            fillA(kb + 1, cur ^ 1);      // DMA drains under the WMMAs below
            loadB(kb + 1, segN);         // global loads in flight during compute
        }
        v16bf bfrag[4];
        #pragma unroll
        for (int t = 0; t < 4; ++t)
            bfrag[t] = *(const v16bf*)
                &lsBt[cur][((t << 4) + l16) * 32 + (half << 4)];
        #pragma unroll
        for (int r = 0; r < 2; ++r) {
            v16bf af = *(const v16bf*)
                &lsA[cur][((wave << 5) + (r << 4) + l16) * 32 + (half << 4)];
            #pragma unroll
            for (int t = 0; t < 4; ++t)
                acc[r][t] = __builtin_amdgcn_wmma_f32_16x16x32_bf16(
                    false, af, false, bfrag[t], (short)0, acc[r][t], false, false);
        }
        if (kb + 1 < Cd / 32) storeB(segN, cur ^ 1);   // wait_loadcnt lands here
    }
    #pragma unroll
    for (int r = 0; r < 2; ++r)
        #pragma unroll
        for (int t = 0; t < 4; ++t)
            #pragma unroll
            for (int v = 0; v < 8; ++v) {
                int row = m0 + (wave << 5) + (r << 4) + v + (half << 3);
                int col = n0 + (t << 4) + l16;
                u16 val = f32_to_bf16(acc[r][t][v]);
                if (VTRANS) {
                    int bb = row >> 9, kp = row & 511;
                    D[((size_t)bb * Cd + col) * Lq + kp] = val;
                } else {
                    D[(size_t)row * Cd + col] = val;
                }
            }
}

// ---------------------------------------------------------------------------
// Fused attention per (batch, 16-query tile). 256 threads = 8 waves.
// 96 KB dynamic LDS: sQ[32K] + double-buffered K/V slabs[2x32K], phase-
// overlapped with sS/sP. All stages async DMA; all fragments contiguous.
// ---------------------------------------------------------------------------
__global__ __launch_bounds__(256)
void attn_kernel(const u16* __restrict__ Q, const u16* __restrict__ Kmat,
                 const u16* __restrict__ Vt, const int* __restrict__ lens,
                 float* __restrict__ out) {
    extern __shared__ __attribute__((aligned(32))) char smem[];   // 98304 B
    u16*   sQ = (u16*)smem;                        // [16][1024] unpermuted, 32KB
    float* sS = (float*)smem;                      // [16][512] f32 (ph2)
    u16*   sP = (u16*)(smem + 65536);              // [16][512] unpermuted (ph2/3)

    const int tid  = threadIdx.x;
    const int wave = tid >> 5, lane = tid & 31;
    const int half = lane >> 4, l16 = lane & 15;
    const int b = blockIdx.y, q0 = blockIdx.x << 4;
    const int len = lens[b];

    // ---- Q tile -> LDS (async, natural k order)
    {
        const int r0 = tid >> 7, c = (tid & 127) << 3, g = tid & 3;
        const unsigned ldsq =
            lds_off(&sQ[r0 * Cd + (c & ~31) + (g << 3)]);
        const unsigned go = (unsigned)(r0 * (Cd * 2) + (c << 1));
        const u16* qbase = Q + (size_t)(b * Lq + q0) * Cd;
        #pragma unroll
        for (int it = 0; it < 8; ++it)
            async_b128(ldsq + it * 4096, go + it * 4096, qbase);
    }

    // ---- Phase 1: S = Q K^T, K slabs double-buffered at +32K / +64K
    v8f Sacc[4];
    for (int t = 0; t < 4; ++t)
        for (int e = 0; e < 8; ++e) Sacc[t][e] = 0.f;

    const int kr0 = tid >> 2, ka8 = (tid & 3) << 3;
    const unsigned kLds[2] = {
        lds_off(smem + 32768 + (kr0 * 32 + ka8) * 2),
        lds_off(smem + 65536 + (kr0 * 32 + ka8) * 2)};
    const unsigned kgo0 = (unsigned)(kr0 * (Cd * 2) + (ka8 << 1));

    auto fillK = [&](int ac, int buf) {
        const u16* kbase = Kmat + (size_t)b * Lq * Cd + (ac << 5);
        #pragma unroll
        for (int it = 0; it < 8; ++it)
            async_b128(kLds[buf] + it * 4096, kgo0 + it * (64 * Cd * 2), kbase);
    };

    fillK(0, 0);
    for (int ac = 0; ac < 32; ++ac) {
        const int cur = ac & 1;
        wait_async0();
        __syncthreads();
        if (ac + 1 < 32) fillK(ac + 1, cur ^ 1);   // drains under WMMAs

        const u16* sKf = (u16*)(smem + 32768 + cur * 32768);
        v16bf af = *(const v16bf*)&sQ[l16 * Cd + (ac << 5) + (half << 4)];
        #pragma unroll
        for (int t = 0; t < 4; ++t) {
            int kt = (wave << 2) + t;
            v16bf bf = *(const v16bf*)&sKf[((kt << 4) + l16) * 32 + (half << 4)];
            Sacc[t] = __builtin_amdgcn_wmma_f32_16x16x32_bf16(
                false, af, false, bf, (short)0, Sacc[t], false, false);
        }
    }
    __syncthreads();

    // ---- masked scores -> LDS (over sQ)
    #pragma unroll
    for (int t = 0; t < 4; ++t) {
        int kt = (wave << 2) + t;
        #pragma unroll
        for (int v = 0; v < 8; ++v) {
            int row = v + (half << 3);
            int kp  = (kt << 4) + l16;
            float s = Sacc[t][v];
            if (kp >= len) s = -2147483648.0f;   // NEG_BIG
            sS[row * Lq + kp] = s;
        }
    }
    __syncthreads();

    // ---- Phase 2: softmax; wave w handles rows 2w, 2w+1. P stored unpermuted.
    int pcol = (((lane >> 3) & 3) << 3) | (lane & 7);
    for (int rr = 0; rr < 2; ++rr) {
        int r = (wave << 1) + rr;
        float m = -3.4e38f;
        #pragma unroll
        for (int j = 0; j < 16; ++j) m = fmaxf(m, sS[r * Lq + lane + j * 32]);
        for (int off = 16; off; off >>= 1) m = fmaxf(m, __shfl_xor(m, off, 32));
        float sum = 0.f;
        #pragma unroll
        for (int j = 0; j < 16; ++j) sum += __expf(sS[r * Lq + lane + j * 32] - m);
        for (int off = 16; off; off >>= 1) sum += __shfl_xor(sum, off, 32);
        float inv = 1.f / sum;
        #pragma unroll
        for (int j = 0; j < 16; ++j)
            sP[r * Lq + j * 32 + pcol] =
                f32_to_bf16(__expf(sS[r * Lq + lane + j * 32] - m) * inv);
    }

    // ---- Phase 3: O = P V + V. V slabs double-buffered at +32K / +0.
    const int vc0 = tid >> 2, vc8 = (tid & 3) << 3;
    const unsigned vLds[2] = {
        lds_off(smem + 32768 + (vc0 * 32 + vc8) * 2),
        lds_off(smem + (vc0 * 32 + vc8) * 2)};
    const unsigned vgo0 = (unsigned)(vc0 * (Lq * 2) + (vc8 << 1));

    auto fillV = [&](int nh, int kc, int buf) {
        const u16* vbase = Vt + ((size_t)b * Cd + (nh << 9)) * Lq + (kc << 5);
        #pragma unroll
        for (int it = 0; it < 8; ++it)
            async_b128(vLds[buf] + it * 4096, vgo0 + it * (64 * Lq * 2), vbase);
    };

    for (int nh = 0; nh < 2; ++nh) {
        v8f Oacc[4];
        for (int t = 0; t < 4; ++t)
            for (int e = 0; e < 8; ++e) Oacc[t][e] = 0.f;

        fillV(nh, 0, 0);
        for (int kc = 0; kc < 16; ++kc) {
            const int cur = kc & 1;
            wait_async0();
            __syncthreads();
            if (kc + 1 < 16) fillV(nh, kc + 1, cur ^ 1);

            const u16* sVt = (u16*)(cur ? smem : smem + 32768);
            v16bf af = *(const v16bf*)&sP[l16 * Lq + (kc << 5) + (half << 4)];
            #pragma unroll
            for (int t = 0; t < 4; ++t) {
                int ncol = (wave << 6) + (t << 4) + l16;
                v16bf bf = *(const v16bf*)&sVt[ncol * 32 + (half << 4)];
                Oacc[t] = __builtin_amdgcn_wmma_f32_16x16x32_bf16(
                    false, af, false, bf, (short)0, Oacc[t], false, false);
            }
        }
        __syncthreads();   // all reads of this nh's buffers done before refill
        #pragma unroll
        for (int t = 0; t < 4; ++t)
            #pragma unroll
            for (int v = 0; v < 8; ++v) {
                int row = v + (half << 3);
                int col = (nh << 9) + (wave << 6) + (t << 4) + l16;
                size_t gi = (size_t)(b * Lq + q0 + row) * Cd + col;
                float res = bf16_to_f32(Vt[((size_t)b * Cd + col) * Lq + q0 + row]);
                out[gi] = Oacc[t][v] + res;          // residual +V
            }
    }
}

extern "C" void kernel_launch(void* const* d_in, const int* in_sizes, int n_in,
                              void* d_out, int out_size, void* d_ws, size_t ws_size,
                              hipStream_t stream) {
    (void)in_sizes; (void)n_in; (void)out_size; (void)ws_size;
    const float* X    = (const float*)d_in[0];
    const int*   lens = (const int*)d_in[1];
    const float* Wq   = (const float*)d_in[2];
    const float* Wk   = (const float*)d_in[3];
    const float* Wv   = (const float*)d_in[4];
    float* out = (float*)d_out;

    u16* Xb  = (u16*)d_ws;                       // [16384][1024] bf16
    u16* Wqb = Xb  + (size_t)Md * Cd;
    u16* Wkb = Wqb + (size_t)Cd * Cd;
    u16* Wvb = Wkb + (size_t)Cd * Cd;
    u16* Qb  = Wvb + (size_t)Cd * Cd;
    u16* Kb  = Qb  + (size_t)Md * Cd;
    u16* Vtb = Kb  + (size_t)Md * Cd;            // V stored [b][col][kpos]

    int nX = Md * Cd, nW = Cd * Cd;
    convert_f32_bf16<<<(nX / 4 + 255) / 256, 256, 0, stream>>>(X,  Xb,  nX);
    convert_f32_bf16<<<(nW / 4 + 255) / 256, 256, 0, stream>>>(Wq, Wqb, nW);
    convert_f32_bf16<<<(nW / 4 + 255) / 256, 256, 0, stream>>>(Wk, Wkb, nW);
    convert_f32_bf16<<<(nW / 4 + 255) / 256, 256, 0, stream>>>(Wv, Wvb, nW);

    dim3 ggrid(Cd / 64, Md / 128);               // 16 x 128 blocks
    gemm_bf16<0><<<ggrid, 128, 0, stream>>>(Xb, Wqb, Qb);
    gemm_bf16<0><<<ggrid, 128, 0, stream>>>(Xb, Wkb, Kb);
    gemm_bf16<1><<<ggrid, 128, 0, stream>>>(Xb, Wvb, Vtb);

    attn_kernel<<<dim3(Lq / 16, Bn), 256, 98304, stream>>>(Qb, Kb, Vtb, lens, out);
}
